// ResiduePairEncoder_62397284876359
// MI455X (gfx1250) — matrix operand order repaired and checked by reading, not compiled, not run on hardware
//
#include <hip/hip_runtime.h>
#include <hip/hip_bf16.h>

#define FEAT_DIM 128
#define MAX_ATOMS 15
#define MAX_AA 22
#define MAX_RELPOS 32
#define AP2 225            // MAX_ATOMS * MAX_ATOMS
#define KD1 256            // 225 padded to mult of 32
#define KO1 416            // 410 padded to mult of 32

// LDS row strides (shorts), padded +8 so dword-stride % 64 is 4 or 20:
// 264 shorts = 132 dw (132%64=4), 424 shorts = 212 dw (212%64=20),
// 136 shorts = 68 dw (68%64=4)  -> conflict-free ds_load_b128 across 16 rows.
#define SAPAD 264
#define SFPAD 424
#define SHPAD 136

typedef __bf16 v16bf __attribute__((ext_vector_type(16)));
typedef float  v8f   __attribute__((ext_vector_type(8)));

union Frag16 { v16bf v; unsigned u[8]; uint4 q[2]; };

__device__ __forceinline__ unsigned short f2bf(float f) {
    unsigned u = __float_as_uint(f);
    u += 0x7FFFu + ((u >> 16) & 1u);     // round-to-nearest-even
    return (unsigned short)(u >> 16);
}

// ---------------------------------------------------------------------------
// Pre-pass: convert a [Korig x 128] fp32 weight matrix to bf16 packed in
// fragment-contiguous B layout. For col-tile t (of 8), kstep ks (of KS),
// lane l (= half*16 + n), dword v:
//   Wp[(((t*KS + ks)*32 + l)*8 + v] = pack(W[ks*32 + half*16 + 2v  ][t*16+n],
//                                          W[ks*32 + half*16 + 2v+1][t*16+n])
// => each lane loads its 8 B dwords as two coalesced b128 loads.
// ---------------------------------------------------------------------------
__global__ void pack_weights(const float* __restrict__ W, unsigned* __restrict__ Wp,
                             int Korig, int KS) {
    int total = 8 * KS * 256;   // tiles * ksteps * (32 lanes * 8 dwords)
    for (int idx = blockIdx.x * blockDim.x + threadIdx.x; idx < total;
         idx += gridDim.x * blockDim.x) {
        int v    = idx & 7;
        int lane = (idx >> 3) & 31;
        int rest = idx >> 8;
        int ks   = rest % KS;
        int t    = rest / KS;
        int half = lane >> 4, n = lane & 15;
        int k0  = ks * 32 + half * 16 + 2 * v;
        int col = t * 16 + n;
        float f0 = (k0     < Korig) ? W[(long)k0 * FEAT_DIM + col]       : 0.f;
        float f1 = (k0 + 1 < Korig) ? W[(long)(k0 + 1) * FEAT_DIM + col] : 0.f;
        Wp[idx] = (unsigned)f2bf(f0) | ((unsigned)f2bf(f1) << 16);
    }
}

// ---------------------------------------------------------------------------
// Geometry helpers
// ---------------------------------------------------------------------------
struct V3 { float x, y, z; };
__device__ __forceinline__ V3 sub3(V3 a, V3 b) { return {a.x-b.x, a.y-b.y, a.z-b.z}; }
__device__ __forceinline__ V3 cross3(V3 a, V3 b) {
    return {a.y*b.z - a.z*b.y, a.z*b.x - a.x*b.z, a.x*b.y - a.y*b.x};
}
__device__ __forceinline__ float dot3(V3 a, V3 b) { return a.x*b.x + a.y*b.y + a.z*b.z; }
__device__ __forceinline__ V3 scale3(V3 a, float s) { return {a.x*s, a.y*s, a.z*s}; }

__device__ __forceinline__ float dihedral(V3 p0, V3 p1, V3 p2, V3 p3) {
    V3 v0 = sub3(p2, p1), v1 = sub3(p0, p1), v2 = sub3(p3, p2);
    V3 u1 = cross3(v0, v1), u2 = cross3(v0, v2);
    float r1 = rsqrtf(dot3(u1, u1) + 1e-10f);
    float r2 = rsqrtf(dot3(u2, u2) + 1e-10f);
    V3 n1 = scale3(u1, r1), n2 = scale3(u2, r2);
    float s = dot3(cross3(v0, n1), n2);
    float sgn = (s > 0.f) ? 1.f : ((s < 0.f) ? -1.f : 0.f);
    float c = fminf(fmaxf(dot3(n1, n2), -1.f + 1e-7f), 1.f - 1e-7f);
    float ang = sgn * acosf(c);
    return isnan(ang) ? 0.f : ang;
}

__device__ __forceinline__ V3 ldv3(const float* __restrict__ p, long res, int atom) {
    const float* q = p + (res * MAX_ATOMS + atom) * 3;
    return {q[0], q[1], q[2]};
}

// ---------------------------------------------------------------------------
// One wave computes TWO 16x16 output tiles (rows 0-15 and 16-31) against one
// shared B fragment per kstep:  C[2][16x16] += A[2](16x32) @ B(32x16).
// A layout (16-bit A 16x32): per lane {half=lane>>4, m=lane&15},
// dwords 0-7 = K 0..15 (+16*half, +32*ks), matching packed B.
// ---------------------------------------------------------------------------
template <int KS>
__device__ __forceinline__ void wmma_rowtile2(const unsigned short* __restrict__ sMat,
                                              int strideShorts,
                                              const unsigned* __restrict__ Wp,
                                              int tile, int lane,
                                              v8f& acc0, v8f& acc1) {
    const int n = lane & 15, half = lane >> 4;
    const int dwStride = strideShorts >> 1;
    const unsigned* sDW = (const unsigned*)sMat;
#pragma unroll
    for (int ks = 0; ks < KS; ++ks) {
        Frag16 a0, a1, b;
        const uint4* bp = (const uint4*)(Wp + (((long)tile * KS + ks) * 32 + lane) * 8);
        b.q[0] = bp[0];
        b.q[1] = bp[1];
        int base0 = n * dwStride + 8 * half + 16 * ks;
        int base1 = (n + 16) * dwStride + 8 * half + 16 * ks;
#pragma unroll
        for (int v = 0; v < 4; ++v) {
            a0.u[v]     = sDW[base0 + v];
            a0.u[4 + v] = sDW[base0 + 4 + v];
            a1.u[v]     = sDW[base1 + v];
            a1.u[4 + v] = sDW[base1 + 4 + v];
        }
        acc0 = __builtin_amdgcn_wmma_f32_16x16x32_bf16(false, a0.v, false, b.v,
                                                       (short)0, acc0, false, false);
        acc1 = __builtin_amdgcn_wmma_f32_16x16x32_bf16(false, a1.v, false, b.v,
                                                       (short)0, acc1, false, false);
    }
}

__device__ __forceinline__ void relu_store_bf16(v8f acc0, v8f acc1,
                                                const float* __restrict__ bias,
                                                unsigned short* __restrict__ dst,
                                                int strideShorts, int colBase, int lane) {
    const int n = lane & 15, half = lane >> 4;
    const float bv = bias[colBase + n];
#pragma unroll
    for (int r = 0; r < 8; ++r) {
        int row = half * 8 + r;
        dst[row * strideShorts + colBase + n]        = f2bf(fmaxf(acc0[r] + bv, 0.f));
        dst[(row + 16) * strideShorts + colBase + n] = f2bf(fmaxf(acc1[r] + bv, 0.f));
    }
}

// ---------------------------------------------------------------------------
// Main kernel: one 256-thread block (8 wave32) per 32 residue pairs.
// ---------------------------------------------------------------------------
__global__ __launch_bounds__(256) void pair_encoder(
    const int* __restrict__ aa, const int* __restrict__ res_nb,
    const int* __restrict__ chain_nb, const float* __restrict__ pos,
    const float* __restrict__ mask_atoms,
    const float* __restrict__ aa_embed, const float* __restrict__ relpos_embed,
    const float* __restrict__ distcoef,
    const float* __restrict__ dist_b1, const float* __restrict__ dist_b2,
    const float* __restrict__ out_b1, const float* __restrict__ out_b2,
    const float* __restrict__ out_b3,
    const unsigned* __restrict__ Wd1p, const unsigned* __restrict__ Wd2p,
    const unsigned* __restrict__ Wo1p, const unsigned* __restrict__ Wo2p,
    const unsigned* __restrict__ Wo3p,
    float* __restrict__ out, int Lres) {

    __shared__ __align__(16) unsigned short sA[32][SAPAD];    // d_gauss bf16
    __shared__ __align__(16) unsigned short sFeat[32][SFPAD]; // [aap|rel|dist|dihed|pad]
    __shared__ __align__(16) unsigned short sH[32][SHPAD];
    __shared__ __align__(16) unsigned short sH2[32][SHPAD];
    __shared__ float sMask[32];

    const int tid = threadIdx.x;
    const int m = tid >> 3, l8 = tid & 7;   // 8 threads per pair, 32 pairs

    // ---- Stage 1: build features for the block's 32 pairs ----
    {
        long pr = (long)blockIdx.x * 32 + m;
        long LL = (long)Lres * Lres;
        int b = (int)(pr / LL);
        int rem = (int)(pr % LL);
        int i = rem / Lres, j = rem % Lres;
        long ri = (long)b * Lres + i, rj = (long)b * Lres + j;

        int ap = aa[ri] * MAX_AA + aa[rj];
        bool same = (chain_nb[ri] == chain_nb[rj]);
        int rel = res_nb[ri] - res_nb[rj];
        rel = min(max(rel, -MAX_RELPOS), MAX_RELPOS) + MAX_RELPOS;

        for (int c = l8; c < FEAT_DIM; c += 8) {
            sFeat[m][c] = f2bf(aa_embed[(long)ap * FEAT_DIM + c]);
            float rp = same ? relpos_embed[(long)rel * FEAT_DIM + c] : 0.f;
            sFeat[m][FEAT_DIM + c] = f2bf(rp);
        }

        for (int k = l8; k < AP2; k += 8) {
            int a1 = k / MAX_ATOMS, a2 = k % MAX_ATOMS;
            const float* pi = pos + (ri * MAX_ATOMS + a1) * 3;
            const float* pj = pos + (rj * MAX_ATOMS + a2) * 3;
            float dx = pi[0] - pj[0], dy = pi[1] - pj[1], dz = pi[2] - pj[2];
            float d = sqrtf(dx * dx + dy * dy + dz * dz + 1e-10f) * 0.1f;
            float cw = distcoef[(long)ap * AP2 + k];
            float cc = (cw > 20.f) ? cw : log1pf(__expf(cw));     // softplus
            float ma = mask_atoms[ri * MAX_ATOMS + a1] * mask_atoms[rj * MAX_ATOMS + a2];
            sA[m][k] = f2bf(__expf(-cc * d * d) * ma);
        }
        for (int k = AP2 + l8; k < KD1; k += 8) sA[m][k] = 0;

        if (l8 == 1) {
            for (int c = 3 * FEAT_DIM + 26; c < KO1; ++c) sFeat[m][c] = 0;
        }
        if (l8 == 0) {
            sMask[m] = mask_atoms[ri * MAX_ATOMS + 1] * mask_atoms[rj * MAX_ATOMS + 1];
            V3 Ni = ldv3(pos, ri, 0), CAi = ldv3(pos, ri, 1), Ci = ldv3(pos, ri, 2);
            V3 Nj = ldv3(pos, rj, 0), CAj = ldv3(pos, rj, 1), Cj = ldv3(pos, rj, 2);
            float phi = dihedral(Ci, Nj, CAj, Cj);
            float psi = dihedral(Ni, CAi, Ci, Nj);
            const float fr[6] = {1.f, 2.f, 3.f, 1.f, 0.5f, 1.f / 3.f};
            float angs[2] = {phi, psi};
#pragma unroll
            for (int q = 0; q < 2; ++q) {
                int base = 3 * FEAT_DIM + q * 13;
                sFeat[m][base] = f2bf(angs[q]);
#pragma unroll
                for (int f = 0; f < 6; ++f) {
                    sFeat[m][base + 1 + f] = f2bf(__sinf(angs[q] * fr[f]));
                    sFeat[m][base + 7 + f] = f2bf(__cosf(angs[q] * fr[f]));
                }
            }
        }
    }
    __syncthreads();

    // ---- GEMM stages: 8 waves; wave w owns columns [w*16, w*16+16), all 32 rows
    const int lane = tid & 31;
    const int tile = tid >> 5;
    const int colBase = tile * 16;
    v8f acc0, acc1;

    // dist MLP layer 1: relu(d_gauss[32x225] @ dist_w1 + b1)
    acc0 = (v8f){}; acc1 = (v8f){};
    wmma_rowtile2<8>(&sA[0][0], SAPAD, Wd1p, tile, lane, acc0, acc1);
    relu_store_bf16(acc0, acc1, dist_b1, &sH[0][0], SHPAD, colBase, lane);
    __syncthreads();

    // dist MLP layer 2 -> feat_all[:, 256:384]
    acc0 = (v8f){}; acc1 = (v8f){};
    wmma_rowtile2<4>(&sH[0][0], SHPAD, Wd2p, tile, lane, acc0, acc1);
    relu_store_bf16(acc0, acc1, dist_b2, &sFeat[0][2 * FEAT_DIM], SFPAD, colBase, lane);
    __syncthreads();

    // out MLP layer 1: relu(feat_all[32x410] @ out_w1 + b1)
    acc0 = (v8f){}; acc1 = (v8f){};
    wmma_rowtile2<13>(&sFeat[0][0], SFPAD, Wo1p, tile, lane, acc0, acc1);
    relu_store_bf16(acc0, acc1, out_b1, &sH[0][0], SHPAD, colBase, lane);
    __syncthreads();

    // out MLP layer 2
    acc0 = (v8f){}; acc1 = (v8f){};
    wmma_rowtile2<4>(&sH[0][0], SHPAD, Wo2p, tile, lane, acc0, acc1);
    relu_store_bf16(acc0, acc1, out_b2, &sH2[0][0], SHPAD, colBase, lane);
    __syncthreads();

    // out MLP layer 3 + mask_pair, fp32 store
    acc0 = (v8f){}; acc1 = (v8f){};
    wmma_rowtile2<4>(&sH2[0][0], SHPAD, Wo3p, tile, lane, acc0, acc1);
    {
        const int n = lane & 15, half = lane >> 4;
        const float bv = out_b3[colBase + n];
        long rowBase = (long)blockIdx.x * 32;
#pragma unroll
        for (int r = 0; r < 8; ++r) {
            int row = half * 8 + r;
            out[(rowBase + row) * FEAT_DIM + colBase + n] =
                (acc0[r] + bv) * sMask[row];
            out[(rowBase + row + 16) * FEAT_DIM + colBase + n] =
                (acc1[r] + bv) * sMask[row + 16];
        }
    }
}

// ---------------------------------------------------------------------------
// Launch
// ---------------------------------------------------------------------------
extern "C" void kernel_launch(void* const* d_in, const int* in_sizes, int n_in,
                              void* d_out, int out_size, void* d_ws, size_t ws_size,
                              hipStream_t stream) {
    const int*   aa         = (const int*)d_in[0];
    const int*   res_nb     = (const int*)d_in[1];
    const int*   chain_nb   = (const int*)d_in[2];
    const float* pos_atoms  = (const float*)d_in[3];
    const float* mask_atoms = (const float*)d_in[4];
    const float* aa_embed   = (const float*)d_in[5];
    const float* relpos_w   = (const float*)d_in[6];
    const float* distcoef   = (const float*)d_in[7];
    const float* dist_w1    = (const float*)d_in[8];
    const float* dist_b1    = (const float*)d_in[9];
    const float* dist_w2    = (const float*)d_in[10];
    const float* dist_b2    = (const float*)d_in[11];
    const float* out_w1     = (const float*)d_in[12];
    const float* out_b1     = (const float*)d_in[13];
    const float* out_w2     = (const float*)d_in[14];
    const float* out_b2     = (const float*)d_in[15];
    const float* out_w3     = (const float*)d_in[16];
    const float* out_b3     = (const float*)d_in[17];
    float* out = (float*)d_out;

    const int NL = in_sizes[0];                 // N * L
    const long P = (long)out_size / FEAT_DIM;   // N * L * L
    const int L = (int)(P / NL);

    // Packed bf16 weights in workspace (dword offsets)
    unsigned* ws   = (unsigned*)d_ws;
    unsigned* Wd1p = ws;                        // 8*8*256  = 16384 dwords
    unsigned* Wd2p = ws + 16384;                // 8*4*256  =  8192
    unsigned* Wo1p = ws + 24576;                // 8*13*256 = 26624
    unsigned* Wo2p = ws + 51200;                //  8192
    unsigned* Wo3p = ws + 59392;                //  8192
    (void)ws_size; (void)n_in;

    pack_weights<<<64, 256, 0, stream>>>(dist_w1, Wd1p, AP2, KD1 / 32);
    pack_weights<<<32, 256, 0, stream>>>(dist_w2, Wd2p, FEAT_DIM, FEAT_DIM / 32);
    pack_weights<<<104, 256, 0, stream>>>(out_w1, Wo1p, 3 * FEAT_DIM + 26, KO1 / 32);
    pack_weights<<<32, 256, 0, stream>>>(out_w2, Wo2p, FEAT_DIM, FEAT_DIM / 32);
    pack_weights<<<32, 256, 0, stream>>>(out_w3, Wo3p, FEAT_DIM, FEAT_DIM / 32);

    int blocks = (int)(P / 32);
    pair_encoder<<<blocks, 256, 0, stream>>>(
        aa, res_nb, chain_nb, pos_atoms, mask_atoms,
        aa_embed, relpos_w, distcoef,
        dist_b1, dist_b2, out_b1, out_b2, out_b3,
        Wd1p, Wd2p, Wo1p, Wo2p, Wo3p,
        out, L);
}
